// Qwen3Attention_17308718202895
// MI455X (gfx1250) — matrix operand order repaired and checked by reading, not compiled, not run on hardware
//
#include <hip/hip_runtime.h>
#include <hip/hip_bf16.h>

// ---------------- problem constants ----------------
enum : int { T = 4096, HID = 2048, H = 16, KV = 8, D = 128 };
enum : int { NQ = H * D, NKV = KV * D };                 // 2048, 1024
#define SCALE 0.08838834764831845f                       // 128^-0.5
#define LOG_THETA 13.815510557964274f                    // ln(1e6)

// ---------------- WMMA types ----------------
typedef __attribute__((ext_vector_type(16))) __bf16 v16bf;
typedef __attribute__((ext_vector_type(8)))  float  v8f;

union FragBF {
    v16bf v;
    uint4 u[2];
};

__device__ inline v8f wmma_bf16(const FragBF& a, const FragBF& b, v8f c) {
    return __builtin_amdgcn_wmma_f32_16x16x32_bf16(false, a.v, false, b.v,
                                                   (short)0, c, false, false);
}

__device__ inline unsigned short f2bf(float f) {
    unsigned int u = __float_as_uint(f);
    u += 0x7fffu + ((u >> 16) & 1u);     // round-to-nearest-even
    return (unsigned short)(u >> 16);
}

// Async global -> LDS DMA (ASYNCcnt path, CDNA5 ISA).
// laddr = workgroup-relative LDS byte address (low 32 bits of flat shared ptr).
__device__ inline void async_ld_b128(unsigned laddr, const void* gaddr) {
    asm volatile("global_load_async_to_lds_b128 %0, %1, off"
                 :: "v"(laddr), "v"(gaddr) : "memory");
}
__device__ inline void wait_async0() {
    asm volatile("s_wait_asynccnt 0" ::: "memory");
}

// ---------------- DPP16 butterfly reductions over 16-lane rows ----------------
// xor1 = quad_perm[1,0,3,2] (0xB1), xor2 = quad_perm[2,3,0,1] (0x4E),
// then row_half_mirror (0x141) == xor4 and row_mirror (0x140) == xor8 once
// the smaller groups are uniform. Pure VALU, no LDS pipe.
template <int CTRL>
__device__ inline float dpp16(float x) {
    int i = __float_as_int(x);
    return __int_as_float(__builtin_amdgcn_update_dpp(i, i, CTRL, 0xF, 0xF, true));
}
__device__ inline float row_max16(float x) {
    x = fmaxf(x, dpp16<0xB1>(x));
    x = fmaxf(x, dpp16<0x4E>(x));
    x = fmaxf(x, dpp16<0x141>(x));
    x = fmaxf(x, dpp16<0x140>(x));
    return x;
}
__device__ inline float row_sum16(float x) {
    x += dpp16<0xB1>(x);
    x += dpp16<0x4E>(x);
    x += dpp16<0x141>(x);
    x += dpp16<0x140>(x);
    return x;
}

// ---------------- f32 -> bf16 conversion ----------------
__global__ void cvt_f32_bf16(const float* __restrict__ in,
                             unsigned short* __restrict__ out, int n) {
    int i = blockIdx.x * blockDim.x + threadIdx.x;
    if (i < n) out[i] = f2bf(in[i]);
}

// ---------------- bf16 WMMA GEMM: C[M,N] = A[M,K] @ B[K,N] ----------------
// Compile-time N,K so store/load strides fold into immediate offsets.
// Block tile 128x64, BK=32, 128 threads = 4 waves; wave w owns rows [w*32, w*32+32).
template <int N, int K>
__global__ void __launch_bounds__(128) gemm_bf16(const unsigned short* __restrict__ A,
                                                 const unsigned short* __restrict__ B,
                                                 float* __restrict__ C) {
    __shared__ __align__(16) unsigned short lA[128][32];   // 8 KB
    __shared__ __align__(16) unsigned short lBt[64][32];   // 4 KB (B transposed)

    const int tid  = threadIdx.x;
    const int lane = tid & 31;
    const int wave = tid >> 5;        // 0..3
    const int hl   = lane >> 4;       // half-wave select
    const int l16  = lane & 15;
    const int m0   = blockIdx.y * 128;
    const int n0   = blockIdx.x * 64;

    v8f acc[2][4];
#pragma unroll
    for (int s = 0; s < 2; ++s)
#pragma unroll
        for (int c = 0; c < 4; ++c)
#pragma unroll
            for (int e = 0; e < 8; ++e) acc[s][c][e] = 0.0f;

    for (int k0 = 0; k0 < K; k0 += 32) {
        // --- A tile 128x32 via async DMA: thread t stages row t (64 B) ---
        {
            const unsigned short* g = A + (size_t)(m0 + tid) * K + k0;
            unsigned la = (unsigned)(size_t)&lA[tid][0];
#pragma unroll
            for (int q = 0; q < 4; ++q)
                async_ld_b128(la + q * 16, g + q * 8);
        }
        // --- B tile staged transposed: lBt[n][k] = B[k0+k][n0+n] ---
        {
            int k   = tid >> 2;             // 0..31
            int n16 = (tid & 3) * 16;       // 0,16,32,48
            const uint4* src = (const uint4*)(B + (size_t)(k0 + k) * N + n0 + n16);
            uint4 d0 = src[0], d1 = src[1];
            const unsigned short* p0 = (const unsigned short*)&d0;
            const unsigned short* p1 = (const unsigned short*)&d1;
#pragma unroll
            for (int j = 0; j < 8; ++j) lBt[n16 + j][k] = p0[j];
#pragma unroll
            for (int j = 0; j < 8; ++j) lBt[n16 + 8 + j][k] = p1[j];
        }
        wait_async0();
        __syncthreads();

        // A fragments (16x32): e<8 -> k = hl*8+e ; e>=8 -> k = 16+hl*8+(e-8)
        FragBF a[2];
#pragma unroll
        for (int s = 0; s < 2; ++s) {
            a[s].u[0] = *(const uint4*)&lA[wave * 32 + s * 16 + l16][hl * 8];
            a[s].u[1] = *(const uint4*)&lA[wave * 32 + s * 16 + l16][16 + hl * 8];
        }
#pragma unroll
        for (int c = 0; c < 4; ++c) {
            // B fragment (32x16): n = l16, e<8 -> k = hl*8+e ; e>=8 -> k = 16+hl*8+(e-8)
            FragBF b;
            b.u[0] = *(const uint4*)&lBt[c * 16 + l16][hl * 8];
            b.u[1] = *(const uint4*)&lBt[c * 16 + l16][16 + hl * 8];
            acc[0][c] = wmma_bf16(a[0], b, acc[0][c]);
            acc[1][c] = wmma_bf16(a[1], b, acc[1][c]);
        }
        __syncthreads();
    }

    // C/D layout: VGPR e -> row = e + hl*8, col = l16; strides compile-time
#pragma unroll
    for (int s = 0; s < 2; ++s) {
        float* base = C + (size_t)(m0 + wave * 32 + s * 16 + hl * 8) * N + n0 + l16;
#pragma unroll
        for (int c = 0; c < 4; ++c)
#pragma unroll
            for (int e = 0; e < 8; ++e)
                base[(size_t)e * N + c * 16] = acc[s][c][e];
    }
}

// ---------------- RMSNorm + RoPE + bf16 convert, relayout [T,HH,D] -> [HH,T,D] ----
__global__ void __launch_bounds__(128) norm_rope_cvt(const float* __restrict__ in,
                                                     unsigned short* __restrict__ out,
                                                     const float* __restrict__ w,
                                                     const int* __restrict__ pos,
                                                     int HH, int do_rope) {
    int t = blockIdx.x;
    int h = blockIdx.y;
    int i = threadIdx.x;          // 0..127

    __shared__ float xs[128];
    __shared__ float red[4];

    float x = in[((size_t)t * HH + h) * D + i];
    if (w != nullptr) {
        float ss = x * x;
        ss += __shfl_xor(ss, 1, 32);
        ss += __shfl_xor(ss, 2, 32);
        ss += __shfl_xor(ss, 4, 32);
        ss += __shfl_xor(ss, 8, 32);
        ss += __shfl_xor(ss, 16, 32);
        if ((i & 31) == 0) red[i >> 5] = ss;
        __syncthreads();
        float tot = red[0] + red[1] + red[2] + red[3];
        x = x * rsqrtf(tot * (1.0f / 128.0f) + 1e-6f) * w[i];
    }
    xs[i] = x;
    __syncthreads();

    float r = x;
    if (do_rope) {
        int   hi  = i & 63;
        float inv = __expf(-(float)hi * (LOG_THETA / 64.0f));
        float ang = (float)pos[t] * inv;
        float c, s;
        __sincosf(ang, &c, &s);
        float x1 = xs[hi], x2 = xs[hi + 64];
        r = (i < 64) ? (x1 * c - x2 * s) : (x2 * c + x1 * s);
    }
    out[((size_t)h * T + t) * D + i] = f2bf(r);
}

// ---------------- flash-style causal GQA attention --------------------
// grid: (T/64, H), 128 threads (4 waves). Wave w owns q rows [q0+16w, q0+16w+16).
__global__ void __launch_bounds__(128) attn_kernel(const unsigned short* __restrict__ Qb,
                                                   const unsigned short* __restrict__ Kb,
                                                   const unsigned short* __restrict__ Vb,
                                                   float* __restrict__ O) {
    const int h    = blockIdx.y;
    const int kvh  = h >> 1;                 // GQA rep = 2
    const int q0   = blockIdx.x * 64;
    const int tid  = threadIdx.x;
    const int wave = tid >> 5;
    const int lane = tid & 31;
    const int hl   = lane >> 4;
    const int l16  = lane & 15;
    const int qb   = q0 + wave * 16;

    __shared__ __align__(16) unsigned short lK[32][128];    // 8 KB
    __shared__ __align__(16) unsigned short lVt[128][32];   // 8 KB, V transposed
    __shared__ __align__(16) unsigned short lP[4][16][32];  // 4 KB, per-wave P

    // Q fragments for D=128 (4 chunks of 32)
    FragBF qf[4];
    const unsigned short* qrow = Qb + ((size_t)h * T + qb + l16) * D;
#pragma unroll
    for (int kk = 0; kk < 4; ++kk) {
        qf[kk].u[0] = *(const uint4*)(qrow + kk * 32 + hl * 8);
        qf[kk].u[1] = *(const uint4*)(qrow + kk * 32 + 16 + hl * 8);
    }

    v8f o[8];
#pragma unroll
    for (int c = 0; c < 8; ++c)
#pragma unroll
        for (int e = 0; e < 8; ++e) o[c][e] = 0.0f;
    float mrow[8], lrow[8];
#pragma unroll
    for (int e = 0; e < 8; ++e) { mrow[e] = -1e30f; lrow[e] = 0.0f; }

    const int nchunks = q0 / 32 + 2;   // keys up to q0+63 inclusive

    for (int jc = 0; jc < nchunks; ++jc) {
        const int kbase = jc * 32;
        // --- K chunk [32][128] via async DMA; V^T [128][32] via scatter ---
        {
            int row = tid >> 2;            // 0..31
            int seg = (tid & 3) * 32;      // 0,32,64,96
            const unsigned short* gk = Kb + ((size_t)kvh * T + kbase + row) * D + seg;
            unsigned la = (unsigned)(size_t)&lK[row][seg];
#pragma unroll
            for (int q = 0; q < 4; ++q)
                async_ld_b128(la + q * 16, gk + q * 8);

            const uint4* vs = (const uint4*)(Vb + ((size_t)kvh * T + kbase + row) * D + seg);
#pragma unroll
            for (int q4 = 0; q4 < 4; ++q4) {
                uint4 dv = vs[q4];
                const unsigned short* dd = (const unsigned short*)&dv;
#pragma unroll
                for (int j = 0; j < 8; ++j) lVt[seg + q4 * 8 + j][row] = dd[j];
            }
        }
        wait_async0();
        __syncthreads();

        if (kbase <= qb + 15) {           // wave-uniform causal skip
            // S = Q @ K^T for two 16-key subtiles
            v8f s[2];
#pragma unroll
            for (int sub = 0; sub < 2; ++sub) {
                v8f sa;
#pragma unroll
                for (int e = 0; e < 8; ++e) sa[e] = 0.0f;
#pragma unroll
                for (int kk = 0; kk < 4; ++kk) {
                    FragBF b;   // B 32x16: n=key=l16(+sub*16), k = hl*8+e / 16+hl*8+(e-8) over d
                    b.u[0] = *(const uint4*)&lK[sub * 16 + l16][kk * 32 + hl * 8];
                    b.u[1] = *(const uint4*)&lK[sub * 16 + l16][kk * 32 + 16 + hl * 8];
                    sa = wmma_bf16(qf[kk], b, sa);
                }
                s[sub] = sa;
            }

            // scale + causal mask + online softmax; row reductions via DPP16
#pragma unroll
            for (int e = 0; e < 8; ++e) {
                int   r  = qb + hl * 8 + e;
                float s0 = s[0][e] * SCALE;
                float s1 = s[1][e] * SCALE;
                if (kbase + l16 > r)      s0 = -1e30f;
                if (kbase + 16 + l16 > r) s1 = -1e30f;

                float mx = row_max16(fmaxf(s0, s1));
                float mn = fmaxf(mrow[e], mx);

                float p0 = __expf(s0 - mn);
                float p1 = __expf(s1 - mn);
                float rs = row_sum16(p0 + p1);

                float corr = __expf(mrow[e] - mn);
                lrow[e] = lrow[e] * corr + rs;
                mrow[e] = mn;
#pragma unroll
                for (int c = 0; c < 8; ++c) o[c][e] *= corr;

                // C-layout -> A-layout via per-wave LDS scratch
                lP[wave][hl * 8 + e][l16]      = f2bf(p0);
                lP[wave][hl * 8 + e][16 + l16] = f2bf(p1);
            }
            asm volatile("s_wait_dscnt 0" ::: "memory");   // wave-local LDS visibility

            // O += P @ V  (P: 16x32 A-frag; V^T gives contiguous B-frags)
            FragBF pf;
            pf.u[0] = *(const uint4*)&lP[wave][l16][hl * 8];
            pf.u[1] = *(const uint4*)&lP[wave][l16][16 + hl * 8];
#pragma unroll
            for (int c = 0; c < 8; ++c) {
                FragBF vf;  // n = d = c*16+l16, k = key = hl*8+e / 16+hl*8+(e-8)
                vf.u[0] = *(const uint4*)&lVt[c * 16 + l16][hl * 8];
                vf.u[1] = *(const uint4*)&lVt[c * 16 + l16][16 + hl * 8];
                o[c] = wmma_bf16(pf, vf, o[c]);
            }
        }
        __syncthreads();
    }

    // normalize + write O[T][H*D]
#pragma unroll
    for (int e = 0; e < 8; ++e) {
        float inv = 1.0f / lrow[e];
        float* base = O + (size_t)(qb + hl * 8 + e) * NQ + h * D + l16;
#pragma unroll
        for (int c = 0; c < 8; ++c)
            base[c * 16] = o[c][e] * inv;
    }
}

// ---------------- host-side launch ----------------
extern "C" void kernel_launch(void* const* d_in, const int* in_sizes, int n_in,
                              void* d_out, int out_size, void* d_ws, size_t ws_size,
                              hipStream_t stream) {
    const float* hid  = (const float*)d_in[0];
    const int*   pos  = (const int*)d_in[1];
    const float* wq   = (const float*)d_in[2];
    const float* wk   = (const float*)d_in[3];
    const float* wv   = (const float*)d_in[4];
    const float* wo   = (const float*)d_in[5];
    const float* qnw  = (const float*)d_in[6];
    const float* knw  = (const float*)d_in[7];
    float* out = (float*)d_out;

    // workspace carve-out
    char* ws = (char*)d_ws;
    size_t off = 0;
    auto carve = [&](size_t bytes) {
        void* p = ws + off;
        off += (bytes + 255) & ~(size_t)255;
        return p;
    };
    unsigned short* hid_bf = (unsigned short*)carve((size_t)T * HID * 2);
    unsigned short* wq_bf  = (unsigned short*)carve((size_t)HID * NQ * 2);
    unsigned short* wk_bf  = (unsigned short*)carve((size_t)HID * NKV * 2);
    unsigned short* wv_bf  = (unsigned short*)carve((size_t)HID * NKV * 2);
    unsigned short* wo_bf  = (unsigned short*)carve((size_t)NQ * HID * 2);
    float*          qf     = (float*)carve((size_t)T * NQ * 4);
    float*          kf     = (float*)carve((size_t)T * NKV * 4);
    float*          vf     = (float*)carve((size_t)T * NKV * 4);
    unsigned short* qbf    = (unsigned short*)carve((size_t)T * NQ * 2);   // [H][T][D]
    unsigned short* kbf    = (unsigned short*)carve((size_t)T * NKV * 2);  // [KV][T][D]
    unsigned short* vbf    = (unsigned short*)carve((size_t)T * NKV * 2);  // [KV][T][D]
    float*          atn    = (float*)carve((size_t)T * NQ * 4);
    unsigned short* atn_bf = (unsigned short*)carve((size_t)T * NQ * 2);
    (void)ws_size; (void)in_sizes; (void)n_in; (void)out_size;

    // 1) bf16 conversion of streaming operands
    auto cvt = [&](const float* src, unsigned short* dst, int n) {
        cvt_f32_bf16<<<n / 256, 256, 0, stream>>>(src, dst, n);
    };
    cvt(hid, hid_bf, T * HID);
    cvt(wq,  wq_bf,  HID * NQ);
    cvt(wk,  wk_bf,  HID * NKV);
    cvt(wv,  wv_bf,  HID * NKV);
    cvt(wo,  wo_bf,  NQ * HID);

    // 2) QKV projections (M = T rows, 128-row block tiles)
    gemm_bf16<NQ,  HID><<<dim3(NQ / 64,  T / 128), 128, 0, stream>>>(hid_bf, wq_bf, qf);
    gemm_bf16<NKV, HID><<<dim3(NKV / 64, T / 128), 128, 0, stream>>>(hid_bf, wk_bf, kf);
    gemm_bf16<NKV, HID><<<dim3(NKV / 64, T / 128), 128, 0, stream>>>(hid_bf, wv_bf, vf);

    // 3) RMSNorm + RoPE + convert + head-major relayout
    norm_rope_cvt<<<dim3(T, H),  128, 0, stream>>>(qf, qbf, qnw,     pos, H,  1);
    norm_rope_cvt<<<dim3(T, KV), 128, 0, stream>>>(kf, kbf, knw,     pos, KV, 1);
    norm_rope_cvt<<<dim3(T, KV), 128, 0, stream>>>(vf, vbf, nullptr, pos, KV, 0);

    // 4) causal GQA flash attention
    attn_kernel<<<dim3(T / 64, H), 128, 0, stream>>>(qbf, kbf, vbf, atn);

    // 5) output projection
    cvt(atn, atn_bf, T * NQ);
    gemm_bf16<HID, NQ><<<dim3(HID / 64, T / 128), 128, 0, stream>>>(atn_bf, wo_bf, out);
}
